// MultiScaleRetention_35330400977592
// MI455X (gfx1250) — matrix operand order repaired and checked by reading, not compiled, not run on hardware
//
#include <hip/hip_runtime.h>
#include <cstdint>
#include <cstddef>

typedef __bf16 bf16;
typedef __attribute__((ext_vector_type(16))) __bf16 v16bf;
typedef __attribute__((ext_vector_type(8)))  __bf16 v8bf;
typedef __attribute__((ext_vector_type(8)))  float  v8f;

union V16 { v16bf v; v8bf h[2]; };

static constexpr int B_ = 2, T_ = 2048, E_ = 2048, H_ = 8, DK_ = 256;
static constexpr int M_ = B_ * T_;          // 4096 token rows
static constexpr float KSCALE = 0.0625f;    // DK^-0.5
static constexpr float EPS_ = 1e-6f;

// ---------------------------------------------------------------------------
// WMMA helpers. Fragment layouts per CDNA5 ISA 7.12.2 (wave32):
//  A 16x32 bf16 : lane holds row M=lane%16; 16 values = two contiguous
//                 8-element K-chunks at k0+8*hi and k0+16+8*hi (hi=lane/16).
//  B 32x16 bf16 : lane holds col N=n0+lane%16; same two 8-element K-chunks
//                 at k0+8*hi and k0+16+8*hi.  (B source stored N-major, K contig.)
//  C/D 16x16 f32: VGPR r -> row r+8*hi, col = lane%16.
// ---------------------------------------------------------------------------
__device__ __forceinline__ v8f wmma_acc(v16bf a, v16bf b, v8f c) {
  return __builtin_amdgcn_wmma_f32_16x16x32_bf16(
      /*neg_a=*/false, a, /*neg_b=*/false, b,
      /*c_mod=*/(short)0, c, /*reuse_a=*/false, /*reuse_b=*/false);
}

__device__ __forceinline__ v16bf load_a_frag(const bf16* A, int lda,
                                             int m0, int k0, int lane) {
  int m  = m0 + (lane & 15);
  int hi = (lane >> 4) & 1;
  const bf16* p = A + (size_t)m * lda + k0 + 8 * hi;
  V16 u;
  u.h[0] = *(const v8bf*)(p);        // K = k0+8*hi .. +7
  u.h[1] = *(const v8bf*)(p + 16);   // K = k0+16+8*hi .. +7
  return u.v;
}

// BT is stored N-major: BT[n][k], ldb = K stride of a row.
__device__ __forceinline__ v16bf load_bt_frag(const bf16* BT, int ldb,
                                              int n0, int k0, int lane) {
  int n  = n0 + (lane & 15);
  int hi = (lane >> 4) & 1;
  const bf16* p = BT + (size_t)n * ldb + k0 + 8 * hi;
  V16 u;
  u.h[0] = *(const v8bf*)(p);        // K = k0+8*hi .. +7
  u.h[1] = *(const v8bf*)(p + 16);   // K = k0+16+8*hi .. +7
  return u.v;
}

// ---------------------------------------------------------------------------
// gfx1250 async global->LDS copy engine (ASYNCcnt). Inline asm so the same
// source works on both ROCm-7.2 and amdgpu-toolchain (their TDM/async
// builtins differ in arity). The LDS address operand is the low 32 bits of
// the generic pointer, matching the hardware's own flat->LDS mapping
// (ISA 10.2).
// ---------------------------------------------------------------------------
__device__ __forceinline__ void async_ld_b128(const bf16* lds_dst,
                                              const bf16* gsrc) {
  unsigned lds_off = (unsigned)(uintptr_t)lds_dst;
  unsigned long long ga = (unsigned long long)(uintptr_t)gsrc;
  asm volatile("global_load_async_to_lds_b128 %0, %1, off"
               :: "v"(lds_off), "v"(ga) : "memory");
}
__device__ __forceinline__ void wait_async0() {
  asm volatile("s_wait_asynccnt 0" ::: "memory");
}

// ---------------------------------------------------------------------------
// Packing kernels
// ---------------------------------------------------------------------------
__global__ void cvt_f32_bf16(const float* __restrict__ src,
                             bf16* __restrict__ dst, int n) {
  int i = blockIdx.x * blockDim.x + threadIdx.x;
  if (i < n) dst[i] = (bf16)src[i];
}

// W[k][n] (E x E, fp32) -> WT[n][k] (bf16)
__global__ void transpose_w_bf16(const float* __restrict__ W,
                                 bf16* __restrict__ WT) {
  int i = blockIdx.x * blockDim.x + threadIdx.x;
  if (i >= E_ * E_) return;
  int n = i % E_, k = i / E_;
  WT[(size_t)n * E_ + k] = (bf16)W[(size_t)k * E_ + n];
}

// RoPE (theta shift) + head-major repack:
//  P[(b*T+t)][h*DK+d] (fp32) -> dst[((b*H+h)*T+t)*DK+d] (bf16), k also scaled.
__global__ void rope_pack(const float* __restrict__ P,
                          const float* __restrict__ snt,
                          const float* __restrict__ cst,
                          bf16* __restrict__ dst, float scale) {
  int i = blockIdx.x * blockDim.x + threadIdx.x;  // over B*T*H*DK/2 pairs
  if (i >= B_ * T_ * E_ / 2) return;
  int dp = i % (DK_ / 2);
  int h  = (i / (DK_ / 2)) % H_;
  int t  = (i / (E_ / 2)) % T_;
  int b  = i / (T_ * E_ / 2);
  int d  = 2 * dp;
  size_t src = (size_t)(b * T_ + t) * E_ + h * DK_ + d;
  float x1 = P[src] * scale, x2 = P[src + 1] * scale;
  float c = cst[t * DK_ + d], s = snt[t * DK_ + d];
  size_t o = ((size_t)(b * H_ + h) * T_ + t) * DK_ + d;
  dst[o]     = (bf16)(x1 * c - x2 * s);   // even: x*cos - x[odd]*sin
  dst[o + 1] = (bf16)(x2 * c + x1 * s);   // odd : x*cos + x[even]*sin
}

// V[(b*T+t)][h*DK+d] fp32 -> VT[((b*H+h)*DK+d)*T+t] bf16  (d-major for B-frags)
__global__ void pack_vt(const float* __restrict__ V, bf16* __restrict__ VT) {
  int i = blockIdx.x * blockDim.x + threadIdx.x;
  if (i >= B_ * T_ * E_) return;
  int d = i % DK_;
  int h = (i / DK_) % H_;
  int t = (i / E_) % T_;
  int b = i / (T_ * E_);
  VT[((size_t)(b * H_ + h) * DK_ + d) * T_ + t] = (bf16)V[i];
}

// ---------------------------------------------------------------------------
// bf16 WMMA GEMM:  C[M][N] (f32) = A[M][K] (bf16) x BT[N][K] (bf16, N-major)
// Block tile 128x128 (8 waves, 4x2), wave tile 32x64 (8 WMMA per K-step).
// Double-buffered 128x32 LDS panels filled by the gfx1250 async global->LDS
// engine: iteration i issues the copy of panel i+1, computes WMMAs on the
// resident panel i, and only then drains ASYNCcnt -> copy/compute overlap.
// ---------------------------------------------------------------------------
__global__ __launch_bounds__(256)
void gemm_bf16_tn(const bf16* __restrict__ A, const bf16* __restrict__ BT,
                  float* __restrict__ C, int M, int N, int K) {
  __shared__ __align__(16) bf16 sA[2][128 * 32];
  __shared__ __align__(16) bf16 sB[2][128 * 32];
  int tid  = threadIdx.x;
  int lane = tid & 31;
  int wave = tid >> 5;                   // 0..7
  int wm = wave >> 1, wn = wave & 1;     // 4 x 2 wave grid
  int bm = blockIdx.y * 128;
  int bn = blockIdx.x * 128;

  // Cooperative staging map: thread -> (row 0..127, 16-elem segment 0/16)
  int crow = tid >> 1;
  int cseg = (tid & 1) * 16;
  const bf16* gA = A  + (size_t)(bm + crow) * K + cseg;
  const bf16* gB = BT + (size_t)(bn + crow) * K + cseg;
  bf16* lA[2] = {&sA[0][crow * 32 + cseg], &sA[1][crow * 32 + cseg]};
  bf16* lB[2] = {&sB[0][crow * 32 + cseg], &sB[1][crow * 32 + cseg]};

  // Prologue: stage panel 0.
  async_ld_b128(lA[0],     gA);
  async_ld_b128(lA[0] + 8, gA + 8);
  async_ld_b128(lB[0],     gB);
  async_ld_b128(lB[0] + 8, gB + 8);
  wait_async0();
  __syncthreads();

  v8f acc[2][4] = {};
  int cur = 0;
  for (int k0 = 0; k0 < K; k0 += 32) {
    int nxt = cur ^ 1;
    if (k0 + 32 < K) {   // issue next panel; overlaps the WMMAs below
      async_ld_b128(lA[nxt],     gA + k0 + 32);
      async_ld_b128(lA[nxt] + 8, gA + k0 + 40);
      async_ld_b128(lB[nxt],     gB + k0 + 32);
      async_ld_b128(lB[nxt] + 8, gB + k0 + 40);
      if (k0 + 288 < K) {  // global_prefetch_b8, 8 panels ahead
        __builtin_prefetch(gA + k0 + 288, 0, 1);
        __builtin_prefetch(gB + k0 + 288, 0, 1);
      }
    }

    v16bf a0 = load_a_frag(sA[cur], 32, wm * 32,      0, lane);
    v16bf a1 = load_a_frag(sA[cur], 32, wm * 32 + 16, 0, lane);
#pragma unroll
    for (int j = 0; j < 4; j++) {
      v16bf b = load_bt_frag(sB[cur], 32, wn * 64 + j * 16, 0, lane);
      acc[0][j] = wmma_acc(a0, b, acc[0][j]);
      acc[1][j] = wmma_acc(a1, b, acc[1][j]);
    }

    wait_async0();       // next panel resident (in this wave)
    __syncthreads();     // all waves done reading cur / writing nxt
    cur = nxt;
  }

  int col = lane & 15, hi = lane >> 4;
#pragma unroll
  for (int i = 0; i < 2; i++)
#pragma unroll
    for (int j = 0; j < 4; j++)
#pragma unroll
      for (int r = 0; r < 8; r++)
        C[(size_t)(bm + wm * 32 + 16 * i + r + 8 * hi) * N +
          bn + wn * 64 + 16 * j + col] = acc[i][j][r];
}

// ---------------------------------------------------------------------------
// Fused retention attention: per (b, h, 16-row t-block), single pass over s.
//   scores = (qr . kr) * mask ; rowsum accumulated ; acc += scores @ V
//   out = acc / max(|rowsum|, 1)   (denominator is a per-row scalar)
// Causal skip: mask==0 for s>t, so only s0 <= t0+15 blocks are processed.
// ---------------------------------------------------------------------------
__global__ __launch_bounds__(32)
void retention_attn(const bf16* __restrict__ QR, const bf16* __restrict__ KR,
                    const bf16* __restrict__ VT, const float* __restrict__ MASK,
                    float* __restrict__ OUT) {
  __shared__ __align__(16) bf16 sqk[16 * 32];   // masked scores, t x s tile
  __shared__ float srow[16];

  int lane = threadIdx.x & 31;
  int col = lane & 15, hi = lane >> 4;
  int t0 = blockIdx.x * 16;
  int h  = blockIdx.y;
  int b  = blockIdx.z;

  const bf16* qh = QR + (size_t)(b * H_ + h) * T_ * DK_;   // [t][d]
  const bf16* kh = KR + (size_t)(b * H_ + h) * T_ * DK_;   // [s][d] == B-layout
  const bf16* vh = VT + (size_t)(b * H_ + h) * DK_ * T_;   // [d][s] == B-layout
  const float* mh = MASK + (size_t)h * T_ * T_;

  // Preload all 8 A-fragments of qr for this t-block (full DK=256).
  v16bf qf[8];
#pragma unroll
  for (int d = 0; d < 8; d++) qf[d] = load_a_frag(qh, DK_, t0, d * 32, lane);

  v8f acc[16] = {};     // 16 t-rows x 256 d-cols accumulator
  float rsum = 0.f;     // per-row score sum (lanes 0..15 hold row=lane)

  for (int s0 = 0; s0 <= t0 + 15; s0 += 32) {
    // ---- scores: 16x32 tile of qr . kr^T over DK ----
    v8f c0 = {}, c1 = {};
#pragma unroll
    for (int d = 0; d < 8; d++) {
      v16bf b0 = load_bt_frag(kh, DK_, s0,      d * 32, lane);
      v16bf b1 = load_bt_frag(kh, DK_, s0 + 16, d * 32, lane);
      c0 = wmma_acc(qf[d], b0, c0);
      c1 = wmma_acc(qf[d], b1, c1);
    }
    // ---- apply decay mask, stage bf16 tile in LDS (C-layout -> A-layout) ----
#pragma unroll
    for (int r = 0; r < 8; r++) {
      int tt = t0 + r + 8 * hi;
      size_t mrow = (size_t)tt * T_ + s0 + col;
      sqk[(r + 8 * hi) * 32 + col]      = (bf16)(c0[r] * mh[mrow]);
      sqk[(r + 8 * hi) * 32 + col + 16] = (bf16)(c1[r] * mh[mrow + 16]);
    }
    __syncthreads();
    // ---- denominator: per-row sum of masked scores ----
    if (lane < 16) {
      float rs = 0.f;
#pragma unroll
      for (int s = 0; s < 32; s++) rs += (float)sqk[lane * 32 + s];
      rsum += rs;
    }
    // ---- scores @ V : A-frag from LDS, 16 WMMAs across DK ----
    V16 u;
    const bf16* lp = &sqk[(lane & 15) * 32 + 8 * hi];
    u.h[0] = *(const v8bf*)(lp);
    u.h[1] = *(const v8bf*)(lp + 16);
    v16bf aq = u.v;
#pragma unroll
    for (int j = 0; j < 16; j++) {
      v16bf bv = load_bt_frag(vh, T_, j * 16, s0, lane);
      acc[j] = wmma_acc(aq, bv, acc[j]);
    }
    __syncthreads();
  }

  if (lane < 16) srow[lane] = rsum;
  __syncthreads();

#pragma unroll
  for (int r = 0; r < 8; r++) {
    int tt = t0 + r + 8 * hi;
    float denom = fabsf(srow[r + 8 * hi]);
    float inv = 1.f / (denom < 1.f ? 1.f : denom);
    size_t obase = (size_t)(b * T_ + tt) * E_ + h * DK_;
#pragma unroll
    for (int j = 0; j < 16; j++)
      OUT[obase + j * 16 + col] = acc[j][r] * inv;
  }
}

// ---------------------------------------------------------------------------
// Fused per-(b,t,h) RMS-norm over DK + SiLU(g) gating -> bf16 for final GEMM.
// ---------------------------------------------------------------------------
__global__ __launch_bounds__(256)
void rms_silu(const float* __restrict__ OATT, const float* __restrict__ G,
              bf16* __restrict__ Y) {
  __shared__ float red[256];
  int tid = threadIdx.x;
  int r = blockIdx.x;                  // over B*T*H
  int h = r % H_;
  int t = (r / H_) % T_;
  int b = r / (H_ * T_);
  size_t base = (size_t)(b * T_ + t) * E_ + h * DK_;
  float x = OATT[base + tid];
  red[tid] = x * x;
  __syncthreads();
  for (int s = 128; s > 0; s >>= 1) {
    if (tid < s) red[tid] += red[tid + s];
    __syncthreads();
  }
  float inv = rsqrtf(red[0] / (float)DK_ + EPS_);
  float g = G[base + tid];
  float silu = g / (1.f + __expf(-g));
  Y[base + tid] = (bf16)(silu * x * inv);
}

// ---------------------------------------------------------------------------
extern "C" void kernel_launch(void* const* d_in, const int* in_sizes, int n_in,
                              void* d_out, int out_size, void* d_ws,
                              size_t ws_size, hipStream_t stream) {
  (void)in_sizes; (void)n_in; (void)out_size; (void)ws_size;
  const float* x    = (const float*)d_in[0];
  const float* snt  = (const float*)d_in[1];
  const float* cst  = (const float*)d_in[2];
  const float* mask = (const float*)d_in[3];
  const float* Wq   = (const float*)d_in[4];
  const float* Wk   = (const float*)d_in[5];
  const float* Wv   = (const float*)d_in[6];
  const float* Wg   = (const float*)d_in[7];
  const float* Wo   = (const float*)d_in[8];
  float* out = (float*)d_out;

  char* ws = (char*)d_ws;
  size_t off = 0;
  auto alloc = [&](size_t bytes) -> char* {
    char* p = ws + off;
    off += (bytes + 255) & ~(size_t)255;
    return p;
  };

  bf16*  xb  = (bf16*)alloc((size_t)M_ * E_ * 2);
  bf16*  wqt = (bf16*)alloc((size_t)E_ * E_ * 2);
  bf16*  wkt = (bf16*)alloc((size_t)E_ * E_ * 2);
  bf16*  wvt = (bf16*)alloc((size_t)E_ * E_ * 2);
  bf16*  wgt = (bf16*)alloc((size_t)E_ * E_ * 2);
  bf16*  wot = (bf16*)alloc((size_t)E_ * E_ * 2);
  float* qf  = (float*)alloc((size_t)M_ * E_ * 4);
  float* kf  = (float*)alloc((size_t)M_ * E_ * 4);
  float* vf  = (float*)alloc((size_t)M_ * E_ * 4);
  float* gf  = (float*)alloc((size_t)M_ * E_ * 4);
  bf16*  qr  = (bf16*)alloc((size_t)M_ * E_ * 2);
  bf16*  kr  = (bf16*)alloc((size_t)M_ * E_ * 2);
  bf16*  vt  = (bf16*)alloc((size_t)M_ * E_ * 2);
  float* oat = (float*)alloc((size_t)M_ * E_ * 4);
  bf16*  y   = (bf16*)alloc((size_t)M_ * E_ * 2);

  const int TPB = 256;
  const int nME = M_ * E_;       // 8,388,608
  const int nEE = E_ * E_;       // 4,194,304

  cvt_f32_bf16<<<nME / TPB, TPB, 0, stream>>>(x, xb, nME);
  transpose_w_bf16<<<nEE / TPB, TPB, 0, stream>>>(Wq, wqt);
  transpose_w_bf16<<<nEE / TPB, TPB, 0, stream>>>(Wk, wkt);
  transpose_w_bf16<<<nEE / TPB, TPB, 0, stream>>>(Wv, wvt);
  transpose_w_bf16<<<nEE / TPB, TPB, 0, stream>>>(Wg, wgt);
  transpose_w_bf16<<<nEE / TPB, TPB, 0, stream>>>(Wo, wot);

  dim3 gg(E_ / 128, M_ / 128);  // (16, 32)
  gemm_bf16_tn<<<gg, 256, 0, stream>>>(xb, wqt, qf, M_, E_, E_);
  gemm_bf16_tn<<<gg, 256, 0, stream>>>(xb, wkt, kf, M_, E_, E_);
  gemm_bf16_tn<<<gg, 256, 0, stream>>>(xb, wvt, vf, M_, E_, E_);
  gemm_bf16_tn<<<gg, 256, 0, stream>>>(xb, wgt, gf, M_, E_, E_);

  const int nPairs = M_ * E_ / 2;
  rope_pack<<<nPairs / TPB, TPB, 0, stream>>>(qf, snt, cst, qr, 1.0f);
  rope_pack<<<nPairs / TPB, TPB, 0, stream>>>(kf, snt, cst, kr, KSCALE);
  pack_vt<<<nME / TPB, TPB, 0, stream>>>(vf, vt);

  dim3 ga(T_ / 16, H_, B_);     // (128, 8, 2)
  retention_attn<<<ga, 32, 0, stream>>>(qr, kr, vt, mask, oat);

  rms_silu<<<B_ * T_ * H_, 256, 0, stream>>>(oat, gf, y);

  gemm_bf16_tn<<<gg, 256, 0, stream>>>(y, wot, out, M_, E_, E_);
}
